// FluidMMRec_30150670418090
// MI455X (gfx1250) — compile-verified, matrix operand order checked by reading, not run on hardware
//
#include <hip/hip_runtime.h>
#include <hip/hip_bf16.h>
#include <math.h>

typedef __attribute__((ext_vector_type(16))) __bf16 v16bf;
typedef __attribute__((ext_vector_type(8)))  __bf16 v8bf;
typedef __attribute__((ext_vector_type(4)))  __bf16 v4bf;
typedef __attribute__((ext_vector_type(8)))  float  v8f;

namespace {
constexpr int Bn = 16, Sn = 512, Hn = 16, Dn = 1024, Pn = 768, dh = 64;
constexpr int NTOK  = Bn * Sn;              // 8192 rows per modality
constexpr int NHTOK = Bn * Hn * Sn;         // 131072 router tokens
constexpr size_t QKV_ELEMS = (size_t)NHTOK * dh;  // 8,388,608 per tensor
}

// ---------------------------------------------------------------- WMMA helpers

__device__ __forceinline__ v8f wmma_bf16(v16bf a, v16bf b, v8f c) {
  return __builtin_amdgcn_wmma_f32_16x16x32_bf16(false, a, false, b, (short)0, c,
                                                 false, false);
}

__device__ __forceinline__ v16bf combine16(v8bf a, v8bf b) {
  return __builtin_shufflevector(a, b, 0, 1, 2, 3, 4, 5, 6, 7,
                                 8, 9, 10, 11, 12, 13, 14, 15);
}

// A fragment (16x32 bf16) from row-major [M x K] buffer (ld = row stride, elems).
// Lane L: row (L&15), k-base = k0 + (L>>4)*8; elems 0..7 -> kb+t, 8..15 -> kb+16+t.
__device__ __forceinline__ v16bf load_a_frag(const __bf16* p, int ld, int row0, int k0) {
  int l  = threadIdx.x & 31;
  int r  = row0 + (l & 15);
  int kb = k0 + ((l >> 4) << 3);
  const __bf16* q = p + (size_t)r * ld + kb;
  v8bf vlo = *(const v8bf*)q;
  v8bf vhi = *(const v8bf*)(q + 16);
  return combine16(vlo, vhi);
}

// B fragment (32x16 bf16) where logical B[k][c] is stored K-contiguous: p[c*ld + k].
__device__ __forceinline__ v16bf load_bt_frag(const __bf16* p, int ld, int k0, int col0) {
  int l  = threadIdx.x & 31;
  int c  = col0 + (l & 15);
  int kb = k0 + ((l >> 4) << 3);
  const __bf16* q = p + (size_t)c * ld + kb;
  v8bf vlo = *(const v8bf*)q;
  v8bf vhi = *(const v8bf*)(q + 16);
  return combine16(vlo, vhi);
}

__device__ __forceinline__ v8bf cvt8(float4 a, float4 b) {
  v8bf r;
  r[0] = (__bf16)a.x; r[1] = (__bf16)a.y; r[2] = (__bf16)a.z; r[3] = (__bf16)a.w;
  r[4] = (__bf16)b.x; r[5] = (__bf16)b.y; r[6] = (__bf16)b.z; r[7] = (__bf16)b.w;
  return r;
}

// ---------------------------------------------------------------- reductions

__device__ __forceinline__ float wred_sum32(float x) {
#pragma unroll
  for (int d = 16; d > 0; d >>= 1) x += __shfl_xor(x, d, 32);
  return x;
}
__device__ __forceinline__ float red_max16(float x) {
#pragma unroll
  for (int d = 8; d > 0; d >>= 1) x = fmaxf(x, __shfl_xor(x, d, 16));
  return x;
}
__device__ __forceinline__ float red_sum16(float x) {
#pragma unroll
  for (int d = 8; d > 0; d >>= 1) x += __shfl_xor(x, d, 16);
  return x;
}

// ---------------------------------------------------------------- staging

// Batched f32 [K x N] -> bf16 transposed [N x K].  Reads coalesced.
__global__ void cvt_transpose_kernel(const float* __restrict__ x, __bf16* __restrict__ y,
                                     int K, int N) {
  const int b = blockIdx.y;
  const int idx = blockIdx.x * 256 + threadIdx.x;
  if (idx < K * N) {
    int k = idx / N, n = idx % N;
    y[(size_t)b * K * N + (size_t)n * K + k] = (__bf16)x[(size_t)b * K * N + idx];
  }
}

// ---------------------------------------------------------------- kernels

// C[8192,1024] = X[8192,768] @ W + bias; W staged transposed [1024][768] bf16.
// Block 256 thr (8 waves): tile M=64, N=256; wave = 32x64 of C.
__global__ void proj_gemm_kernel(const float* __restrict__ img, const float* __restrict__ txt,
                                 const __bf16* __restrict__ wprojT,
                                 const float* __restrict__ img_bias,
                                 const float* __restrict__ txt_bias,
                                 float* __restrict__ projf) {
  __shared__ alignas(16) __bf16 As[64 * 32];
  const int z = blockIdx.z;
  const float*  X    = z ? txt : img;
  const __bf16* WT   = wprojT + (size_t)z * Pn * Dn;   // [1024][768]
  const float*  bias = z ? txt_bias : img_bias;
  const int row0 = blockIdx.x * 64;
  const int col0 = blockIdx.y * 256;
  const int tid  = threadIdx.x;
  const int wave = tid >> 5, lo = tid & 15, lh = (tid >> 4) & 1;
  const int mo = (wave >> 2) * 32, no = (wave & 3) * 64;
  const int sr = tid >> 2, scc = (tid & 3) * 8;        // staging: 8 consecutive elems
  v8f acc[2][4];
#pragma unroll
  for (int a = 0; a < 2; ++a)
#pragma unroll
    for (int b = 0; b < 4; ++b)
#pragma unroll
      for (int j = 0; j < 8; ++j) acc[a][b][j] = 0.f;

  for (int kt = 0; kt < Pn / 32; ++kt) {
    const int kb = kt * 32;
    {   // A: 64x32 f32 -> bf16, vectorized (2x b128 load, 1x b128 LDS store)
      const float* srcp = X + (size_t)(row0 + sr) * Pn + kb + scc;
      float4 f0 = *(const float4*)srcp;
      float4 f1 = *(const float4*)(srcp + 4);
      *(v8bf*)&As[sr * 32 + scc] = cvt8(f0, f1);
    }
    __syncthreads();
    v16bf bfr[4];
#pragma unroll
    for (int nf = 0; nf < 4; ++nf)
      bfr[nf] = load_bt_frag(WT, Pn, kb, col0 + no + nf * 16);
    v16bf a0 = load_a_frag(As, 32, mo, 0);
    v16bf a1 = load_a_frag(As, 32, mo + 16, 0);
#pragma unroll
    for (int nf = 0; nf < 4; ++nf) {
      acc[0][nf] = wmma_bf16(a0, bfr[nf], acc[0][nf]);
      acc[1][nf] = wmma_bf16(a1, bfr[nf], acc[1][nf]);
    }
    __syncthreads();
  }
#pragma unroll
  for (int mi = 0; mi < 2; ++mi)
#pragma unroll
    for (int nf = 0; nf < 4; ++nf) {
      int   c  = col0 + no + nf * 16 + lo;
      float bv = bias[c];
#pragma unroll
      for (int j = 0; j < 8; ++j) {
        int r = row0 + mo + mi * 16 + j + 8 * lh;
        projf[(size_t)z * NTOK * Dn + (size_t)r * Dn + c] = acc[mi][nf][j] + bv;
      }
    }
}

// Per row of 1024: LayerNorm -> L2 normalize -> * d^-0.5, emit bf16 head input.
// Lane handles 4 consecutive elements -> float4 loads, b64 bf16 stores.
__global__ void ln_l2_kernel(const float* __restrict__ proj,
                             const float* __restrict__ img_g, const float* __restrict__ img_beta,
                             const float* __restrict__ txt_g, const float* __restrict__ txt_beta,
                             __bf16* __restrict__ hd) {
  __shared__ float redA[8];
  __shared__ float redB[8];
  const int row = blockIdx.x;             // 0..16383
  const int z = row >> 13, r = row & 8191;
  const float* g  = z ? txt_g : img_g;
  const float* be = z ? txt_beta : img_beta;
  const float* x  = proj + ((size_t)z * NTOK + r) * Dn;
  const int tid = threadIdx.x, lane = tid & 31, wave = tid >> 5;
  const int e0 = tid * 4;
  float4 xv = *(const float4*)(x + e0);
  float v[4] = {xv.x, xv.y, xv.z, xv.w};
  float s = v[0] + v[1] + v[2] + v[3];
  float sq = v[0]*v[0] + v[1]*v[1] + v[2]*v[2] + v[3]*v[3];
  s = wred_sum32(s); sq = wred_sum32(sq);
  if (lane == 0) { redA[wave] = s; redB[wave] = sq; }
  __syncthreads();
  float ts = 0.f, tq = 0.f;
#pragma unroll
  for (int w = 0; w < 8; ++w) { ts += redA[w]; tq += redB[w]; }
  float m = ts * (1.f / 1024.f), var = tq * (1.f / 1024.f) - m * m;
  float ri = rsqrtf(var + 1e-5f);
  float4 gv = *(const float4*)(g + e0);
  float4 bv = *(const float4*)(be + e0);
  float n[4]; float s2 = 0.f;
  n[0] = (v[0]-m)*ri*gv.x + bv.x; n[1] = (v[1]-m)*ri*gv.y + bv.y;
  n[2] = (v[2]-m)*ri*gv.z + bv.z; n[3] = (v[3]-m)*ri*gv.w + bv.w;
#pragma unroll
  for (int i = 0; i < 4; ++i) s2 += n[i] * n[i];
  s2 = wred_sum32(s2);
  __syncthreads();
  if (lane == 0) redA[wave] = s2;
  __syncthreads();
  float t2 = 0.f;
#pragma unroll
  for (int w = 0; w < 8; ++w) t2 += redA[w];
  float sc = 0.125f / fmaxf(sqrtf(t2), 1e-12f);   // L2 + head_dim^-0.5 fold-in
  __bf16* o = hd + ((size_t)z * NTOK + r) * Dn;
  v4bf ov;
#pragma unroll
  for (int i = 0; i < 4; ++i) ov[i] = (__bf16)(n[i] * sc);
  *(v4bf*)&o[e0] = ov;
}

// Per-head QKV projections: 6 tensors x 16 heads; weights transposed [e][d].
__global__ void qkv_kernel(const __bf16* __restrict__ hd, const __bf16* __restrict__ wattnT,
                           const float* __restrict__ bi2tq, const float* __restrict__ bi2tk,
                           const float* __restrict__ bi2tv, const float* __restrict__ bt2iq,
                           const float* __restrict__ bt2ik, const float* __restrict__ bt2iv,
                           __bf16* __restrict__ qkv) {
  __shared__ alignas(16) __bf16 As[64 * 64];
  const int t = blockIdx.z;               // 0..5  (dir*3 + qkv)
  const int dir = t / 3, q = t % 3;
  const int h  = blockIdx.y;
  const int rt = blockIdx.x * 64;
  const int src_mod = (q == 0) ? dir : (1 - dir);
  const __bf16* src = hd + (size_t)src_mod * NTOK * Dn;
  const __bf16* wT  = wattnT + ((size_t)(dir * 4 + q) * Hn + h) * 4096;  // [e][d]
  const float* bias;
  switch (t) {
    case 0: bias = bi2tq; break; case 1: bias = bi2tk; break; case 2: bias = bi2tv; break;
    case 3: bias = bt2iq; break; case 4: bias = bt2ik; break; default: bias = bt2iv; break;
  }
  bias += h * 64;
  const int tid = threadIdx.x;
  const int wave = tid >> 5, lo = tid & 15, lh = (tid >> 4) & 1;
  {   // stage A 64x64 bf16, vectorized: 16 consecutive elems per thread
    const int r = tid >> 2, c = (tid & 3) * 16;
    const __bf16* sp = src + (size_t)(rt + r) * Dn + h * 64 + c;
    *(v8bf*)&As[r * 64 + c]     = *(const v8bf*)sp;
    *(v8bf*)&As[r * 64 + c + 8] = *(const v8bf*)(sp + 8);
  }
  __syncthreads();
  const int mrow = (wave & 3) * 16, nh = (wave >> 2) * 32;
  v16bf bfr[2][2];
#pragma unroll
  for (int ks = 0; ks < 2; ++ks)
#pragma unroll
    for (int nf = 0; nf < 2; ++nf)
      bfr[ks][nf] = load_bt_frag(wT, 64, ks * 32, nh + nf * 16);
  v8f acc[2];
#pragma unroll
  for (int nf = 0; nf < 2; ++nf)
#pragma unroll
    for (int j = 0; j < 8; ++j) acc[nf][j] = 0.f;
#pragma unroll
  for (int ks = 0; ks < 2; ++ks) {
    v16bf a = load_a_frag(As, 64, mrow, ks * 32);
#pragma unroll
    for (int nf = 0; nf < 2; ++nf)
      acc[nf] = wmma_bf16(a, bfr[ks][nf], acc[nf]);
  }
#pragma unroll
  for (int nf = 0; nf < 2; ++nf) {
    int   c  = nh + nf * 16 + lo;
    float bv = bias[c];
#pragma unroll
    for (int j = 0; j < 8; ++j) {
      int r  = rt + mrow + j + 8 * lh;
      int bb = r >> 9, ss = r & 511;
      qkv[(size_t)t * QKV_ELEMS + (((size_t)bb * Hn + h) * Sn + ss) * dh + c] =
          (__bf16)(acc[nf][j] + bv);
    }
  }
}

// Flash attention + output projection + residual -> comb[b,h,s, dir*64 + e].
// Block 128 thr = 4 waves; one wave = 16 queries x full d=64; K/V tiles shared in LDS.
__global__ void attention_kernel(const __bf16* __restrict__ qkv, const __bf16* __restrict__ hd,
                                 const __bf16* __restrict__ wattnT,
                                 const float* __restrict__ bo_i2t, const float* __restrict__ bo_t2i,
                                 float* __restrict__ comb) {
  __shared__ alignas(16) __bf16 Ks[32 * 64];        // [key][d]
  __shared__ alignas(16) __bf16 Vst[64 * 40];       // [d][key], padded ld=40
  __shared__ alignas(16) __bf16 Ps[4][16 * 32];
  __shared__ alignas(16) __bf16 Os[4][16 * 64];
  const int h  = blockIdx.y;
  const int zz = blockIdx.z;
  const int b  = zz & 15, dir = zz >> 4;
  const int tid  = threadIdx.x;
  const int wave = tid >> 5, lo = tid & 15, lh = (tid >> 4) & 1;
  const int qt = blockIdx.x * 4 + wave;
  const size_t bh = ((size_t)b * Hn + h) * Sn;
  const __bf16* Q  = qkv + (size_t)(dir * 3 + 0) * QKV_ELEMS + bh * dh;
  const __bf16* Kp = qkv + (size_t)(dir * 3 + 1) * QKV_ELEMS + bh * dh;
  const __bf16* Vp = qkv + (size_t)(dir * 3 + 2) * QKV_ELEMS + bh * dh;
  const int skey = tid >> 2, sc = (tid & 3) * 16;   // staging: 16 consecutive per thread

  v16bf aq0 = load_a_frag(Q, dh, qt * 16, 0);
  v16bf aq1 = load_a_frag(Q, dh, qt * 16, 32);

  float mrow[8], lrow[8];
  v8f o[4];
  v8f z8;
#pragma unroll
  for (int j = 0; j < 8; ++j) { mrow[j] = -1e30f; lrow[j] = 0.f; z8[j] = 0.f; }
#pragma unroll
  for (int nf = 0; nf < 4; ++nf)
#pragma unroll
    for (int j = 0; j < 8; ++j) o[nf][j] = 0.f;

  for (int kt = 0; kt < Sn / 32; ++kt) {
    const int kb = kt * 32;
    {   // cooperative K/V tile load: K vectorized, V element-scatter transpose
      const __bf16* kp = Kp + (size_t)(kb + skey) * dh + sc;
      v8bf k0 = *(const v8bf*)kp, k1 = *(const v8bf*)(kp + 8);
      *(v8bf*)&Ks[skey * 64 + sc]     = k0;
      *(v8bf*)&Ks[skey * 64 + sc + 8] = k1;
      const __bf16* vp = Vp + (size_t)(kb + skey) * dh + sc;
      v8bf v0 = *(const v8bf*)vp, v1 = *(const v8bf*)(vp + 8);
#pragma unroll
      for (int tt = 0; tt < 8; ++tt) {
        Vst[(sc + tt) * 40 + skey]     = v0[tt];
        Vst[(sc + 8 + tt) * 40 + skey] = v1[tt];
      }
    }
    __syncthreads();

    v16bf bk00 = load_bt_frag(Ks, dh, 0, 0);
    v16bf bk10 = load_bt_frag(Ks, dh, 32, 0);
    v16bf bk01 = load_bt_frag(Ks, dh, 0, 16);
    v16bf bk11 = load_bt_frag(Ks, dh, 32, 16);
    v8f s0 = wmma_bf16(aq0, bk00, z8);
    s0     = wmma_bf16(aq1, bk10, s0);
    v8f s1 = wmma_bf16(aq0, bk01, z8);
    s1     = wmma_bf16(aq1, bk11, s1);

    v16bf bv[4];                        // hoist V frags: LDS latency hides under softmax
#pragma unroll
    for (int nf = 0; nf < 4; ++nf) bv[nf] = load_bt_frag(Vst, 40, 0, nf * 16);

    __bf16* P = Ps[wave];
#pragma unroll
    for (int j = 0; j < 8; ++j) {       // online softmax; row = 16 lanes of a half
      float x0 = s0[j] * 0.125f, x1 = s1[j] * 0.125f;   // scores * d^-0.5
      float mx = red_max16(fmaxf(x0, x1));
      float mn = fmaxf(mrow[j], mx);
      float al = __expf(mrow[j] - mn);
      float p0 = __expf(x0 - mn), p1 = __expf(x1 - mn);
      float rs = red_sum16(p0 + p1);
      lrow[j] = lrow[j] * al + rs;
      mrow[j] = mn;
#pragma unroll
      for (int nf = 0; nf < 4; ++nf) o[nf][j] *= al;
      int pr = (j + 8 * lh) * 32;
      P[pr + lo]      = (__bf16)p0;
      P[pr + 16 + lo] = (__bf16)p1;
    }
    v16bf ap = load_a_frag(P, 32, 0, 0);   // P back in A-fragment layout via LDS
#pragma unroll
    for (int nf = 0; nf < 4; ++nf)
      o[nf] = wmma_bf16(ap, bv[nf], o[nf]);
    __syncthreads();
  }

  // O /= l, transpose through LDS, apply per-head WO + bias + residual.
  __bf16* Ot = Os[wave];
#pragma unroll
  for (int nf = 0; nf < 4; ++nf)
#pragma unroll
    for (int j = 0; j < 8; ++j)
      Ot[(j + 8 * lh) * 64 + nf * 16 + lo] = (__bf16)(o[nf][j] / lrow[j]);
  v16bf ao0 = load_a_frag(Ot, 64, 0, 0);
  v16bf ao1 = load_a_frag(Ot, 64, 0, 32);
  const __bf16* woT = wattnT + ((size_t)(dir * 4 + 3) * Hn + h) * 4096;  // [e][d]
  const float*  bo  = (dir ? bo_t2i : bo_i2t) + h * 64;
  v8f co[4];
#pragma unroll
  for (int nf = 0; nf < 4; ++nf)
#pragma unroll
    for (int j = 0; j < 8; ++j) co[nf][j] = 0.f;
#pragma unroll
  for (int nf = 0; nf < 4; ++nf) {
    co[nf] = wmma_bf16(ao0, load_bt_frag(woT, 64, 0, nf * 16), co[nf]);
    co[nf] = wmma_bf16(ao1, load_bt_frag(woT, 64, 32, nf * 16), co[nf]);
  }
  const __bf16* res = hd + (size_t)dir * NTOK * Dn;   // residual = scaled MHA input
#pragma unroll
  for (int nf = 0; nf < 4; ++nf) {
    int   e  = nf * 16 + lo;
    float bv = bo[e];
#pragma unroll
    for (int j = 0; j < 8; ++j) {
      int   s = qt * 16 + j + 8 * lh;
      float r = (float)res[((size_t)b * Sn + s) * Dn + h * 64 + e];
      comb[(bh + s) * 128 + dir * 64 + e] = co[nf][j] + bv + r;
    }
  }
}

// Per token: head-LN -> out "normalized"; then L2 + router LN1 -> xln (bf16).
// Lane handles 4 consecutive elements -> float4 traffic.
__global__ void headnorm_kernel(const float* __restrict__ comb,
                                const float* __restrict__ hn_g, const float* __restrict__ hn_b,
                                const float* __restrict__ g1, const float* __restrict__ b1,
                                float* __restrict__ out_norm, __bf16* __restrict__ xln) {
  const int tid = threadIdx.x, lane = tid & 31, wave = tid >> 5;
  const int tok = blockIdx.x * 8 + wave;
  const int h   = (tok >> 9) & 15;
  const int e0  = lane * 4;
  const float* x = comb + (size_t)tok * 128;
  float4 xv = *(const float4*)(x + e0);
  float v[4] = {xv.x, xv.y, xv.z, xv.w};
  float s = v[0]+v[1]+v[2]+v[3];
  float sq = v[0]*v[0]+v[1]*v[1]+v[2]*v[2]+v[3]*v[3];
  s = wred_sum32(s); sq = wred_sum32(sq);
  float m = s * (1.f / 128.f), var = sq * (1.f / 128.f) - m * m;
  float ri = rsqrtf(var + 1e-5f);
  float4 gg = *(const float4*)(hn_g + h * 128 + e0);
  float4 bb = *(const float4*)(hn_b + h * 128 + e0);
  float n[4];
  n[0] = (v[0]-m)*ri*gg.x + bb.x; n[1] = (v[1]-m)*ri*gg.y + bb.y;
  n[2] = (v[2]-m)*ri*gg.z + bb.z; n[3] = (v[3]-m)*ri*gg.w + bb.w;
  float s2 = n[0]*n[0]+n[1]*n[1]+n[2]*n[2]+n[3]*n[3];
  float4 onv = {n[0], n[1], n[2], n[3]};
  *(float4*)(out_norm + (size_t)tok * 128 + e0) = onv;
  s2 = wred_sum32(s2);
  float inv = 1.f / fmaxf(sqrtf(s2), 1e-12f);
  float xn[4]; float s3 = 0.f, q3 = 0.f;
#pragma unroll
  for (int i = 0; i < 4; ++i) { xn[i] = n[i] * inv; s3 += xn[i]; q3 += xn[i] * xn[i]; }
  s3 = wred_sum32(s3); q3 = wred_sum32(q3);
  float m1 = s3 * (1.f / 128.f), v1 = q3 * (1.f / 128.f) - m1 * m1;
  float r1 = rsqrtf(v1 + 1e-5f);
  float4 g1v = *(const float4*)(g1 + e0);
  float4 b1v = *(const float4*)(b1 + e0);
  v4bf ov;
  ov[0] = (__bf16)((xn[0]-m1)*r1*g1v.x + b1v.x);
  ov[1] = (__bf16)((xn[1]-m1)*r1*g1v.y + b1v.y);
  ov[2] = (__bf16)((xn[2]-m1)*r1*g1v.z + b1v.z);
  ov[3] = (__bf16)((xn[3]-m1)*r1*g1v.w + b1v.w);
  *(v4bf*)&xln[(size_t)tok * 128 + e0] = ov;
}

// Router: GEMM1 (WMMA, w1 transposed [256][128]) + GeLU + LN2 + expert matvec +
// softmax(T=0.1) + clip.  Block 256 thr = 8 waves, 32 tokens per block.
__global__ void router_kernel(const __bf16* __restrict__ xln, const __bf16* __restrict__ w1T,
                              const float* __restrict__ rb1, const float* __restrict__ g2,
                              const float* __restrict__ be2, const float* __restrict__ w2,
                              const float* __restrict__ rb2, float* __restrict__ gates) {
  __shared__ alignas(16) __bf16 Xs[32 * 128];
  __shared__ alignas(16) float  Hs[32 * 256];
  const int t0  = blockIdx.x * 32;
  const int tid = threadIdx.x, lane = tid & 31, wave = tid >> 5;
  const int lo = tid & 15, lh = (tid >> 4) & 1;
  {   // stage X tile 32x128 bf16, 16 consecutive elems per thread
    const int r = tid >> 3, c = (tid & 7) * 16;
    const __bf16* sp = xln + (size_t)(t0 + r) * 128 + c;
    *(v8bf*)&Xs[r * 128 + c]     = *(const v8bf*)sp;
    *(v8bf*)&Xs[r * 128 + c + 8] = *(const v8bf*)(sp + 8);
  }
  __syncthreads();
  const int m0 = (wave & 1) * 16, n0 = (wave >> 1) * 64;
  v8f acc[4];
#pragma unroll
  for (int nf = 0; nf < 4; ++nf)
#pragma unroll
    for (int j = 0; j < 8; ++j) acc[nf][j] = 0.f;
#pragma unroll
  for (int ks = 0; ks < 4; ++ks) {
    v16bf bfr[4];
#pragma unroll
    for (int nf = 0; nf < 4; ++nf)
      bfr[nf] = load_bt_frag(w1T, 128, ks * 32, n0 + nf * 16);
    v16bf a = load_a_frag(Xs, 128, m0, ks * 32);
#pragma unroll
    for (int nf = 0; nf < 4; ++nf)
      acc[nf] = wmma_bf16(a, bfr[nf], acc[nf]);
  }
#pragma unroll
  for (int nf = 0; nf < 4; ++nf) {
    int   c  = n0 + nf * 16 + lo;
    float bv = rb1[c];
#pragma unroll
    for (int j = 0; j < 8; ++j) {
      int   r  = m0 + j + 8 * lh;
      float xv = acc[nf][j] + bv;
      Hs[r * 256 + c] = 0.5f * xv * (1.f + erff(xv * 0.70710678f));  // exact GeLU
    }
  }
  __syncthreads();
  for (int tt = 0; tt < 4; ++tt) {        // LN2: wave handles 4 tokens
    int tl = wave * 4 + tt;
    const int e0 = lane * 8;              // 8 consecutive per lane
    float4 h0 = *(const float4*)&Hs[tl * 256 + e0];
    float4 h1 = *(const float4*)&Hs[tl * 256 + e0 + 4];
    float hv[8] = {h0.x, h0.y, h0.z, h0.w, h1.x, h1.y, h1.z, h1.w};
    float s = 0.f, sq = 0.f;
#pragma unroll
    for (int k = 0; k < 8; ++k) { s += hv[k]; sq += hv[k] * hv[k]; }
    s = wred_sum32(s); sq = wred_sum32(sq);
    float m = s * (1.f / 256.f), var = sq * (1.f / 256.f) - m * m;
    float ri = rsqrtf(var + 1e-5f);
    float4 g0 = *(const float4*)(g2 + e0), g1v = *(const float4*)(g2 + e0 + 4);
    float4 c0 = *(const float4*)(be2 + e0), c1 = *(const float4*)(be2 + e0 + 4);
    float4 o0 = {(hv[0]-m)*ri*g0.x + c0.x, (hv[1]-m)*ri*g0.y + c0.y,
                 (hv[2]-m)*ri*g0.z + c0.z, (hv[3]-m)*ri*g0.w + c0.w};
    float4 o1 = {(hv[4]-m)*ri*g1v.x + c1.x, (hv[5]-m)*ri*g1v.y + c1.y,
                 (hv[6]-m)*ri*g1v.z + c1.z, (hv[7]-m)*ri*g1v.w + c1.w};
    *(float4*)&Hs[tl * 256 + e0]     = o0;
    *(float4*)&Hs[tl * 256 + e0 + 4] = o1;
  }
  __syncthreads();
  // logits: 32 tokens x 8 experts == 256 threads exactly
  const int tl = tid >> 3, e = tid & 7;
  float a2 = rb2[e];
  for (int k = 0; k < 256; ++k) a2 += Hs[tl * 256 + k] * w2[k * 8 + e];
  float lg = a2 * 10.f;                    // / TEMP
  float mx = lg;
#pragma unroll
  for (int d2 = 4; d2 > 0; d2 >>= 1) mx = fmaxf(mx, __shfl_xor(mx, d2, 8));
  float ex = __expf(lg - mx);
  float sm = ex;
#pragma unroll
  for (int d2 = 4; d2 > 0; d2 >>= 1) sm += __shfl_xor(sm, d2, 8);
  float g = ex / sm;
  gates[(size_t)(t0 + tl) * 8 + e] = fminf(fmaxf(g, 1e-6f), 1.f);
}

// ---------------------------------------------------------------- host launch

extern "C" void kernel_launch(void* const* d_in, const int* in_sizes, int n_in,
                              void* d_out, int out_size, void* d_ws, size_t ws_size,
                              hipStream_t stream) {
  (void)in_sizes; (void)n_in; (void)out_size; (void)ws_size;
  const float* image    = (const float*)d_in[0];
  const float* text     = (const float*)d_in[1];
  const float* img_b    = (const float*)d_in[3];
  const float* img_g    = (const float*)d_in[4];
  const float* img_beta = (const float*)d_in[5];
  const float* txt_b    = (const float*)d_in[7];
  const float* txt_g    = (const float*)d_in[8];
  const float* txt_beta = (const float*)d_in[9];
  const float* i2t_bq = (const float*)d_in[14];
  const float* i2t_bk = (const float*)d_in[15];
  const float* i2t_bv = (const float*)d_in[16];
  const float* i2t_bo = (const float*)d_in[17];
  const float* t2i_bq = (const float*)d_in[22];
  const float* t2i_bk = (const float*)d_in[23];
  const float* t2i_bv = (const float*)d_in[24];
  const float* t2i_bo = (const float*)d_in[25];
  const float* hn_g    = (const float*)d_in[26];
  const float* hn_b    = (const float*)d_in[27];
  const float* r_ln1_g = (const float*)d_in[28];
  const float* r_ln1_b = (const float*)d_in[29];
  const float* r_b1    = (const float*)d_in[31];
  const float* r_ln2_g = (const float*)d_in[32];
  const float* r_ln2_b = (const float*)d_in[33];
  const float* r_w2    = (const float*)d_in[34];
  const float* r_b2    = (const float*)d_in[35];

  char*   ws    = (char*)d_ws;
  float*  projf = (float*)ws;                        // 67,108,864 B; reused as comb
  __bf16* hd    = (__bf16*)(ws + 67108864ull);       // 33,554,432 B
  __bf16* qkvb  = (__bf16*)(ws + 100663296ull);      // 100,663,296 B
  __bf16* wb    = (__bf16*)(ws + 201326592ull);      // 4,259,840 B bf16 weights (transposed)
  __bf16* xln   = (__bf16*)(ws + 205586432ull);      // 33,554,432 B

  struct Job { int idx; size_t off; int K; int N; int batch; };
  const Job jobs[11] = {
    {2, 0, 768, 1024, 1}, {6, 786432, 768, 1024, 1},
    {10, 1572864 + 0 * 65536, 64, 64, 16}, {11, 1572864 + 1 * 65536, 64, 64, 16},
    {12, 1572864 + 2 * 65536, 64, 64, 16}, {13, 1572864 + 3 * 65536, 64, 64, 16},
    {18, 1572864 + 4 * 65536, 64, 64, 16}, {19, 1572864 + 5 * 65536, 64, 64, 16},
    {20, 1572864 + 6 * 65536, 64, 64, 16}, {21, 1572864 + 7 * 65536, 64, 64, 16},
    {30, 2097152, 128, 256, 1}
  };
  for (int j = 0; j < 11; ++j) {
    int per = jobs[j].K * jobs[j].N;
    cvt_transpose_kernel<<<dim3((per + 255) / 256, jobs[j].batch), 256, 0, stream>>>(
        (const float*)d_in[jobs[j].idx], wb + jobs[j].off, jobs[j].K, jobs[j].N);
  }

  proj_gemm_kernel<<<dim3(128, 4, 2), 256, 0, stream>>>(image, text, wb, img_b, txt_b, projf);
  ln_l2_kernel<<<16384, 256, 0, stream>>>(projf, img_g, img_beta, txt_g, txt_beta, hd);
  qkv_kernel<<<dim3(128, 16, 6), 256, 0, stream>>>(hd, wb + 1572864,
      i2t_bq, i2t_bk, i2t_bv, t2i_bq, t2i_bk, t2i_bv, qkvb);
  attention_kernel<<<dim3(8, 16, 32), 128, 0, stream>>>(qkvb, hd, wb + 1572864,
      i2t_bo, t2i_bo, projf /* comb reuses proj buffer */);
  headnorm_kernel<<<16384, 256, 0, stream>>>(projf, hn_g, hn_b, r_ln1_g, r_ln1_b,
      (float*)d_out, xln);
  router_kernel<<<4096, 256, 0, stream>>>(xln, wb + 2097152, r_b1, r_ln2_g, r_ln2_b,
      r_w2, r_b2, (float*)d_out + 16777216ull);
}
